// CrossModalAttentionLayer_42382737277520
// MI455X (gfx1250) — matrix operand order repaired and checked by reading, not compiled, not run on hardware
//
#include <hip/hip_runtime.h>
#include <hip/hip_bf16.h>
#include <math.h>

typedef __attribute__((ext_vector_type(16))) _Float16 v16h;
typedef __attribute__((ext_vector_type(8)))  _Float16 v8h;
typedef __attribute__((ext_vector_type(8)))  float    v8f;

// ---------------------------------------------------------------------------
// Problem constants
// ---------------------------------------------------------------------------
#define BB   4
#define TQ   1024
#define TKK  2048
#define DD   1024
#define HH   16
#define HDD  64

// ---------------------------------------------------------------------------
// WMMA helpers (CDNA5 gfx1250, wave32)
// A 16x32 f16: lane l (r=l&15, hf=l>>4) holds row r, K = hf*16 .. hf*16+15
//              contiguous  -> two v8h loads at +hf*16 and +8+hf*16
// B 32x16 f16: lane l holds column r, K = hf*16 .. hf*16+15 contiguous
// C/D 16x16 f32: VGPR v, lane l -> element (M = v + 8*hf, N = r)
// ---------------------------------------------------------------------------
__device__ __forceinline__ v16h comb16(v8h lo, v8h hi) {
  v16h o;
#pragma unroll
  for (int i = 0; i < 8; ++i) { o[i] = lo[i]; o[i + 8] = hi[i]; }
  return o;
}

// p must already include +hf*16 element offset; reads p[0..15]
__device__ __forceinline__ v16h ld_a(const _Float16* p) {
  return comb16(*(const v8h*)p, *(const v8h*)(p + 8));
}

// p must already include +hf*16 element offset; reads 16 contiguous halves
__device__ __forceinline__ v16h ld_b(const _Float16* p) {
  return comb16(*(const v8h*)p, *(const v8h*)(p + 8));
}

__device__ __forceinline__ v8f wmma16(v16h a, v16h b, v8f c) {
  return __builtin_amdgcn_wmma_f32_16x16x32_f16(false, a, false, b, (short)0, c,
                                                false, false);
}

// ---------------------------------------------------------------------------
// Elementwise f32 -> f16 convert
// ---------------------------------------------------------------------------
__global__ void cvt_f32_f16(const float* __restrict__ in,
                            _Float16* __restrict__ out, size_t n) {
  for (size_t i = (size_t)blockIdx.x * blockDim.x + threadIdx.x; i < n;
       i += (size_t)gridDim.x * blockDim.x)
    out[i] = (_Float16)in[i];
}

__global__ void zero_f32(float* __restrict__ p, size_t n) {
  for (size_t i = (size_t)blockIdx.x * blockDim.x + threadIdx.x; i < n;
       i += (size_t)gridDim.x * blockDim.x)
    p[i] = 0.0f;
}

// V (B,Tk,H,HD) -> Vt (B,H,HD,Tk)
__global__ void vtrans(const _Float16* __restrict__ V,
                       _Float16* __restrict__ Vt) {
  size_t n = (size_t)BB * TKK * HH * HDD;
  for (size_t i = (size_t)blockIdx.x * blockDim.x + threadIdx.x; i < n;
       i += (size_t)gridDim.x * blockDim.x) {
    int d = (int)(i & (HDD - 1));
    size_t t1 = i >> 6;
    int h = (int)(t1 & (HH - 1));
    size_t t2 = t1 >> 4;
    int t = (int)(t2 & (TKK - 1));
    int b = (int)(t2 >> 11);
    Vt[(((size_t)(b * HH + h) * HDD + d) << 11) + t] = V[i];
  }
}

// ---------------------------------------------------------------------------
// Generic GEMM: C(MxN) = A(MxK,f16,row-major) * W(NxK,f16,row-major)^T + bias
// One wave per 32x64 tile (2x4 WMMA accumulators -> 8 WMMA per 6 tile loads).
// mode: 0 -> f16 out, 1 -> GELU(exact)+f16, 2 -> f32 out
// ---------------------------------------------------------------------------
__device__ __forceinline__ void gemm_epi(v8f c, int mb, int nb, int r, int hf,
                                         int N, const float* __restrict__ bias,
                                         _Float16* __restrict__ o16,
                                         float* __restrict__ o32, int mode) {
#pragma unroll
  for (int v = 0; v < 8; ++v) {
    int m = mb + v + 8 * hf;
    int n = nb + r;
    float val = c[v] + bias[n];
    if (mode == 1) val = 0.5f * val * (1.0f + erff(val * 0.70710678118654752f));
    size_t idx = (size_t)m * N + n;
    if (mode == 2) o32[idx] = val;
    else           o16[idx] = (_Float16)val;
  }
}

__global__ void __launch_bounds__(32)
gemm_xwt(const _Float16* __restrict__ A, const _Float16* __restrict__ W,
         const float* __restrict__ bias, _Float16* __restrict__ out16,
         float* __restrict__ out32, int M, int N, int K, int mode) {
  int lane = threadIdx.x & 31;
  int r = lane & 15, hf = lane >> 4;
  int n0 = blockIdx.x * 64;
  int m0 = blockIdx.y * 32;

  const v8f vzero = {};
  v8f c[2][4];
#pragma unroll
  for (int i = 0; i < 2; ++i)
#pragma unroll
    for (int j = 0; j < 4; ++j) c[i][j] = vzero;

  const _Float16* Arow0 = A + (size_t)(m0 + r) * K;
  const _Float16* Arow1 = A + (size_t)(m0 + 16 + r) * K;

  for (int k0 = 0; k0 < K; k0 += 32) {
    // prefetch next k-panel one step ahead (global_prefetch_b8)
    if (k0 + 32 < K) {
      __builtin_prefetch(Arow0 + k0 + 32, 0, 3);
      __builtin_prefetch(W + (size_t)(n0 + r) * K + k0 + 32, 0, 3);
    }
    v16h a[2], b[4];
    a[0] = ld_a(Arow0 + k0 + hf * 16);
    a[1] = ld_a(Arow1 + k0 + hf * 16);
#pragma unroll
    for (int j = 0; j < 4; ++j)
      b[j] = ld_b(W + (size_t)(n0 + 16 * j + r) * K + k0 + hf * 16);
#pragma unroll
    for (int i = 0; i < 2; ++i)
#pragma unroll
      for (int j = 0; j < 4; ++j) c[i][j] = wmma16(a[i], b[j], c[i][j]);
  }
#pragma unroll
  for (int i = 0; i < 2; ++i)
#pragma unroll
    for (int j = 0; j < 4; ++j)
      gemm_epi(c[i][j], m0 + 16 * i, n0 + 16 * j, r, hf, N, bias, out16, out32,
               mode);
}

// ---------------------------------------------------------------------------
// Attention: one wave per (b, h, 16-row q tile).
// Q (B*Tq, D) f16 with head slice at col h*64 ; K same for keys ;
// Vt (B,H,HD,Tk) f16.  Full 16 x Tk score strip staged in 64KB LDS (f16).
// ---------------------------------------------------------------------------
__global__ void __launch_bounds__(32)
attn_k(const _Float16* __restrict__ Q, const _Float16* __restrict__ Km,
       const _Float16* __restrict__ Vt, const unsigned char* __restrict__ mask,
       _Float16* __restrict__ ctx, float* __restrict__ attnw) {
  __shared__ _Float16 sP[16 * TKK];  // 64 KB
  int lane = threadIdx.x & 31;
  int r = lane & 15, hf = lane >> 4;
  int q0 = blockIdx.x * 16;
  int h = blockIdx.y;
  int b = blockIdx.z;

  const _Float16* Qb = Q + (size_t)b * TQ * DD + h * HDD;
  const _Float16* Kb = Km + (size_t)b * TKK * DD + h * HDD;
  const _Float16* Vb = Vt + (size_t)(b * HH + h) * HDD * TKK;

  // Q A-operands for the two 32-wide k-dim chunks of HD=64
  v16h aq0 = ld_a(Qb + (size_t)(q0 + r) * DD + 0 + hf * 16);
  v16h aq1 = ld_a(Qb + (size_t)(q0 + r) * DD + 32 + hf * 16);

  float vmax[8];
#pragma unroll
  for (int v = 0; v < 8; ++v) vmax[v] = -3.0e38f;

  // Pass 1: scores -> LDS, running column-local max
  for (int k0 = 0; k0 < TKK; k0 += 16) {
    if (k0 + 16 < TKK)
      __builtin_prefetch(Kb + (size_t)(k0 + 16 + r) * DD, 0, 3);
    v16h bk0 = ld_b(Kb + (size_t)(k0 + r) * DD + 0 + hf * 16);
    v16h bk1 = ld_b(Kb + (size_t)(k0 + r) * DD + 32 + hf * 16);
    v8f s = {};
    s = wmma16(aq0, bk0, s);
    s = wmma16(aq1, bk1, s);
    bool valid = mask[(size_t)b * TKK + k0 + r] != 0;
#pragma unroll
    for (int v = 0; v < 8; ++v) {
      float sv = s[v] * 0.125f;  // 1/sqrt(64)
      if (!valid) sv = -3.0e38f;
      vmax[v] = fmaxf(vmax[v], sv);
      sP[(v + 8 * hf) * TKK + k0 + r] = (_Float16)sv;  // -> -inf if masked
    }
  }
  __syncthreads();

  // row max: reduce across the 16 lanes of each half (rows differ per half)
#pragma unroll
  for (int off = 1; off < 16; off <<= 1) {
#pragma unroll
    for (int v = 0; v < 8; ++v)
      vmax[v] = fmaxf(vmax[v], __shfl_xor(vmax[v], off, 32));
  }

  // Pass 2: p = exp(s - m), accumulate row sum, overwrite LDS with p
  float lsum[8];
#pragma unroll
  for (int v = 0; v < 8; ++v) lsum[v] = 0.0f;
  for (int k0 = 0; k0 < TKK; k0 += 16) {
#pragma unroll
    for (int v = 0; v < 8; ++v) {
      float sv = (float)sP[(v + 8 * hf) * TKK + k0 + r];
      float p = __expf(sv - vmax[v]);
      lsum[v] += p;
      sP[(v + 8 * hf) * TKK + k0 + r] = (_Float16)p;
    }
  }
  __syncthreads();
#pragma unroll
  for (int off = 1; off < 16; off <<= 1) {
#pragma unroll
    for (int v = 0; v < 8; ++v) lsum[v] += __shfl_xor(lsum[v], off, 32);
  }

  // Pass 3: ctx = P @ V ;  P read from LDS in A-layout, Vt rows as B-operands
  v8f c0 = {}, c1 = {}, c2 = {}, c3 = {};
  for (int k0 = 0; k0 < TKK; k0 += 32) {
    const _Float16* pp = &sP[(size_t)r * TKK + k0 + hf * 16];
    v16h pa = comb16(*(const v8h*)pp, *(const v8h*)(pp + 8));
    v16h b0 = ld_b(Vb + (size_t)(0 + r) * TKK + k0 + hf * 16);
    v16h b1 = ld_b(Vb + (size_t)(16 + r) * TKK + k0 + hf * 16);
    v16h b2 = ld_b(Vb + (size_t)(32 + r) * TKK + k0 + hf * 16);
    v16h b3 = ld_b(Vb + (size_t)(48 + r) * TKK + k0 + hf * 16);
    c0 = wmma16(pa, b0, c0);
    c1 = wmma16(pa, b1, c1);
    c2 = wmma16(pa, b2, c2);
    c3 = wmma16(pa, b3, c3);
  }

  // ctx normalize + store as (B,Tq,H,HD) f16
#pragma unroll
  for (int v = 0; v < 8; ++v) {
    int q = q0 + v + 8 * hf;
    float inv = 1.0f / lsum[v];
    size_t base = ((size_t)(b * TQ + q) * HH + h) * HDD;
    ctx[base + 0 + r]  = (_Float16)(c0[v] * inv);
    ctx[base + 16 + r] = (_Float16)(c1[v] * inv);
    ctx[base + 32 + r] = (_Float16)(c2[v] * inv);
    ctx[base + 48 + r] = (_Float16)(c3[v] * inv);
  }

  // Pass 4: head-mean attention weights (atomic accumulate across heads)
  for (int k0 = 0; k0 < TKK; k0 += 16) {
#pragma unroll
    for (int v = 0; v < 8; ++v) {
      float p = (float)sP[(v + 8 * hf) * TKK + k0 + r];
      float w = p / lsum[v] * (1.0f / 16.0f);
      atomicAdd(attnw + (size_t)(b * TQ + q0 + v + 8 * hf) * TKK + k0 + r, w);
    }
  }
}

// ---------------------------------------------------------------------------
// Fused residual + LayerNorm over rows of 1024; optional f16 copy out
// ---------------------------------------------------------------------------
__global__ void __launch_bounds__(256)
ln_fused(const float* __restrict__ A, const float* __restrict__ Bv,
         const float* __restrict__ g, const float* __restrict__ bt,
         float* __restrict__ out32, _Float16* __restrict__ out16) {
  __shared__ float s1[256], s2[256];
  int row = blockIdx.x, tid = threadIdx.x;
  const float* ap = A + (size_t)row * DD;
  const float* bp = Bv + (size_t)row * DD;
  float vloc[4];
  float sum = 0.f, sq = 0.f;
#pragma unroll
  for (int j = 0; j < 4; ++j) {
    int i = tid + 256 * j;
    float v = ap[i] + bp[i];
    vloc[j] = v;
    sum += v;
    sq += v * v;
  }
  s1[tid] = sum;
  s2[tid] = sq;
  __syncthreads();
  for (int off = 128; off > 0; off >>= 1) {
    if (tid < off) { s1[tid] += s1[tid + off]; s2[tid] += s2[tid + off]; }
    __syncthreads();
  }
  float mu = s1[0] * (1.0f / DD);
  float var = s2[0] * (1.0f / DD) - mu * mu;
  float rs = rsqrtf(var + 1e-5f);
#pragma unroll
  for (int j = 0; j < 4; ++j) {
    int i = tid + 256 * j;
    float y = (vloc[j] - mu) * rs * g[i] + bt[i];
    out32[(size_t)row * DD + i] = y;
    if (out16) out16[(size_t)row * DD + i] = (_Float16)y;
  }
}

// ---------------------------------------------------------------------------
// Host launcher
// ---------------------------------------------------------------------------
extern "C" void kernel_launch(void* const* d_in, const int* in_sizes, int n_in,
                              void* d_out, int out_size, void* d_ws,
                              size_t ws_size, hipStream_t stream) {
  (void)in_sizes; (void)n_in; (void)out_size;
  const float* query = (const float*)d_in[0];
  const float* kv = (const float*)d_in[1];
  const unsigned char* kvmask = (const unsigned char*)d_in[2];
  const float* in_proj_w = (const float*)d_in[3];
  const float* in_proj_b = (const float*)d_in[4];
  const float* out_w = (const float*)d_in[5];
  const float* out_b = (const float*)d_in[6];
  const float* ln1_s = (const float*)d_in[7];
  const float* ln1_b = (const float*)d_in[8];
  const float* ln2_s = (const float*)d_in[9];
  const float* ln2_b = (const float*)d_in[10];
  const float* w1 = (const float*)d_in[11];
  const float* b1 = (const float*)d_in[12];
  const float* w2 = (const float*)d_in[13];
  const float* b2 = (const float*)d_in[14];

  constexpr size_t MB = 1ull << 20;
  if (ws_size < 128 * MB) return;  // need 128 MB scratch
  char* ws = (char*)d_ws;

  // workspace layout with lifetime-based reuse (peak 128 MB)
  _Float16* WI16  = (_Float16*)(ws + 0 * MB);     // 6 MB  in_proj_w f16
  _Float16* WO16  = (_Float16*)(ws + 6 * MB);     // 2 MB  out_w f16
  _Float16* W116  = (_Float16*)(ws + 8 * MB);     // 8 MB  w1 f16
  _Float16* W216  = (_Float16*)(ws + 16 * MB);    // 8 MB  w2 f16
  _Float16* QIN16 = (_Float16*)(ws + 24 * MB);    // 8 MB  query f16 (reused as X16)
  _Float16* KV16  = (_Float16*)(ws + 32 * MB);    // 16 MB kv f16
  _Float16* QP16  = (_Float16*)(ws + 48 * MB);    // 8 MB  Q proj (reused by H16)
  _Float16* KP16  = (_Float16*)(ws + 56 * MB);    // 16 MB K proj
  _Float16* VT16  = (_Float16*)(ws + 72 * MB);    // 16 MB V transposed
  _Float16* V16   = (_Float16*)(ws + 88 * MB);    // 16 MB V proj (temp)
  _Float16* CTX16 = (_Float16*)(ws + 88 * MB);    // 8 MB  ctx (after V dead)
  float*    ATT32 = (float*)(ws + 96 * MB);       // 16 MB attended f32
  float*    FFN32 = (float*)(ws + 80 * MB);       // 16 MB ffn f32 (late)
  float*    X32   = (float*)(ws + 112 * MB);      // 16 MB x f32
  _Float16* X16   = QIN16;                        // reuse
  _Float16* H16   = QP16;                         // 32 MB spans 48..80

  float* xout = (float*)d_out;                    // (B,Tq,D) f32
  float* attnw = xout + (size_t)BB * TQ * DD;     // (B,Tq,Tk) f32

  // 1. convert inputs/weights to f16
  cvt_f32_f16<<<2048, 256, 0, stream>>>(query, QIN16, (size_t)BB * TQ * DD);
  cvt_f32_f16<<<2048, 256, 0, stream>>>(kv, KV16, (size_t)BB * TKK * DD);
  cvt_f32_f16<<<2048, 256, 0, stream>>>(in_proj_w, WI16, (size_t)3 * DD * DD);
  cvt_f32_f16<<<2048, 256, 0, stream>>>(out_w, WO16, (size_t)DD * DD);
  cvt_f32_f16<<<2048, 256, 0, stream>>>(w1, W116, (size_t)4 * DD * DD);
  cvt_f32_f16<<<2048, 256, 0, stream>>>(w2, W216, (size_t)4 * DD * DD);

  // 2. QKV projections
  gemm_xwt<<<dim3(DD / 64, (BB * TQ) / 32), 32, 0, stream>>>(
      QIN16, WI16, in_proj_b, QP16, nullptr, BB * TQ, DD, DD, 0);
  gemm_xwt<<<dim3(DD / 64, (BB * TKK) / 32), 32, 0, stream>>>(
      KV16, WI16 + (size_t)DD * DD, in_proj_b + DD, KP16, nullptr, BB * TKK,
      DD, DD, 0);
  gemm_xwt<<<dim3(DD / 64, (BB * TKK) / 32), 32, 0, stream>>>(
      KV16, WI16 + (size_t)2 * DD * DD, in_proj_b + 2 * DD, V16, nullptr,
      BB * TKK, DD, DD, 0);

  // 3. V -> Vt (B,H,HD,Tk)
  vtrans<<<2048, 256, 0, stream>>>(V16, VT16);

  // 4. attention (+ head-mean attn weights via f32 atomics)
  zero_f32<<<2048, 256, 0, stream>>>(attnw, (size_t)BB * TQ * TKK);
  attn_k<<<dim3(TQ / 16, HH, BB), 32, 0, stream>>>(QP16, KP16, VT16, kvmask,
                                                   CTX16, attnw);

  // 5. output projection (f32)
  gemm_xwt<<<dim3(DD / 64, (BB * TQ) / 32), 32, 0, stream>>>(
      CTX16, WO16, out_b, nullptr, ATT32, BB * TQ, DD, DD, 2);

  // 6. x = LN(query + attended)  (f32 + f16 copies)
  ln_fused<<<BB * TQ, 256, 0, stream>>>(query, ATT32, ln1_s, ln1_b, X32, X16);

  // 7. FFN: h = gelu(x @ w1^T + b1)  -> f16 ; ffn = h @ w2^T + b2 -> f32
  gemm_xwt<<<dim3((4 * DD) / 64, (BB * TQ) / 32), 32, 0, stream>>>(
      X16, W116, b1, H16, nullptr, BB * TQ, 4 * DD, DD, 1);
  gemm_xwt<<<dim3(DD / 64, (BB * TQ) / 32), 32, 0, stream>>>(
      H16, W216, b2, nullptr, FFN32, BB * TQ, DD, 4 * DD, 2);

  // 8. out = LN(x + ffn)
  ln_fused<<<BB * TQ, 256, 0, stream>>>(X32, FFN32, ln2_s, ln2_b, xout,
                                        nullptr);
}
